// QuantumKernelLayer_65481071398429
// MI455X (gfx1250) — matrix operand order repaired and checked by reading, not compiled, not run on hardware
//
#include <hip/hip_runtime.h>
#include <hip/hip_bf16.h>
#include <stdint.h>

// ---------------------------------------------------------------------------
// RBF kernel layer on MI455X (gfx1250, wave32, WMMA):
//   out[n,m] = exp(-(||x_n||^2 + ||p_m||^2 - 2 * <x_n, p_m>))
// BF16 WMMA GEMM core (f32 accumulate), double-buffered LDS with register
// staging, row norms accumulated for free during f32->bf16 staging.
// ---------------------------------------------------------------------------

typedef __attribute__((ext_vector_type(16))) __bf16 v16bf;
typedef __attribute__((ext_vector_type(8)))  __bf16 v8bf;
typedef __attribute__((ext_vector_type(8)))  float  v8f;

#define K_DIM      1024
#define TILE_M     128      // x rows per block
#define TILE_N     128      // prototype rows per block
#define KC         64       // K-chunk = 2 WMMA k-steps of 32
#define NCHUNK     (K_DIM / KC)
#define LDS_STRIDE 72       // bf16 elems per LDS row: 64 data + 8 pad = 144 B
                            // (row*36 dwords mod 64 banks -> conflict-free b128)

union FragBF {
  uint4 u[2];
  v16bf v;
};

// 8x f32 -> 8x bf16 (RNE); backend should pick v_cvt_pk_bf16_f32
__device__ __forceinline__ uint4 cvt8_bf16(const float4& a, const float4& b) {
  v8f f = {a.x, a.y, a.z, a.w, b.x, b.y, b.z, b.w};
  v8bf r = __builtin_convertvector(f, v8bf);
  return __builtin_bit_cast(uint4, r);
}

__device__ __forceinline__ float dot8(const float4& a, const float4& b) {
  return a.x*a.x + a.y*a.y + a.z*a.z + a.w*a.w
       + b.x*b.x + b.y*b.y + b.z*b.z + b.w*b.w;
}

extern "C" __global__ __launch_bounds__(256)
void rbf_gemm_wmma(const float* __restrict__ X, const float* __restrict__ P,
                   float* __restrict__ Out, int Mcols) {
  __shared__ alignas(16) unsigned short As[2][TILE_M * LDS_STRIDE];
  __shared__ alignas(16) unsigned short Bs[2][TILE_N * LDS_STRIDE];
  __shared__ float xsq_s[TILE_M];
  __shared__ float psq_s[TILE_N];

  const int tid  = threadIdx.x;
  const int lane = tid & 31;
  const int wave = tid >> 5;
  const int wm   = wave >> 2;   // wave row 0..1  (64 rows each)
  const int wn   = wave & 3;    // wave col 0..3  (32 cols each)
  const int h    = lane >> 4;   // lane half (WMMA fragment layouts)
  const int r    = lane & 15;

  const int rowBase = blockIdx.y * TILE_M;  // into X
  const int colBase = blockIdx.x * TILE_N;  // into P

  if (tid < TILE_M) { xsq_s[tid] = 0.0f; psq_s[tid] = 0.0f; }

  // staging map: 2 threads per tile row, 32 f32 (128 B) per thread per chunk
  const int ldRow  = tid >> 1;   // 0..127
  const int ldPart = tid & 1;
  const float* xPtr = X + (size_t)(rowBase + ldRow) * K_DIM + ldPart * 32;
  const float* pPtr = P + (size_t)(colBase + ldRow) * K_DIM + ldPart * 32;
  const int stOff = ldRow * LDS_STRIDE + ldPart * 32;

  float accX = 0.0f, accP = 0.0f;   // on-the-fly row norms (f32, full K)

  const v8f vzero = {};
  v8f c[4][2];
  #pragma unroll
  for (int mt = 0; mt < 4; ++mt)
    #pragma unroll
    for (int nt = 0; nt < 2; ++nt)
      c[mt][nt] = vzero;

  float4 ax[8], bx[8];              // register-staged chunk (32+32 f32)

  auto loadChunk = [&](int kc) {
    #pragma unroll
    for (int it = 0; it < 4; ++it) {
      ax[2*it]   = *(const float4*)(xPtr + kc + it * 8);
      ax[2*it+1] = *(const float4*)(xPtr + kc + it * 8 + 4);
      bx[2*it]   = *(const float4*)(pPtr + kc + it * 8);
      bx[2*it+1] = *(const float4*)(pPtr + kc + it * 8 + 4);
    }
  };

  auto stageChunk = [&](int buf) {
    unsigned short* as = &As[buf][stOff];
    unsigned short* bs = &Bs[buf][stOff];
    #pragma unroll
    for (int it = 0; it < 4; ++it) {
      accX += dot8(ax[2*it], ax[2*it+1]);
      accP += dot8(bx[2*it], bx[2*it+1]);
      *(uint4*)(as + it * 8) = cvt8_bf16(ax[2*it], ax[2*it+1]);  // ds_store_b128
      *(uint4*)(bs + it * 8) = cvt8_bf16(bx[2*it], bx[2*it+1]);
    }
  };

  // prologue: stage chunk 0 into buffer 0
  loadChunk(0);
  stageChunk(0);
  __syncthreads();

  #pragma unroll 1
  for (int kc = 0; kc < NCHUNK; ++kc) {
    const int cur = kc & 1;

    // issue next chunk's global loads early; latency hidden by WMMA phase
    if (kc + 1 < NCHUNK) loadChunk((kc + 1) * KC);

    #pragma unroll
    for (int ks = 0; ks < 2; ++ks) {
      // A fragment (16x32 bf16): lane-half h holds K runs [8h,8h+8) and
      // [16+8h,16+8h+8) of row r  -> two b128 loads per tile
      FragBF a[4];
      #pragma unroll
      for (int mt = 0; mt < 4; ++mt) {
        const unsigned short* base =
            &As[cur][(wm * 64 + mt * 16 + r) * LDS_STRIDE + ks * 32];
        a[mt].u[0] = *(const uint4*)(base + h * 8);
        a[mt].u[1] = *(const uint4*)(base + 16 + h * 8);
      }
      // B fragment (32x16 bf16): lane-half h holds K runs [8h,8h+8) and
      // [16+8h,16+8h+8) of column r -> two b128 loads per tile
      FragBF b[2];
      #pragma unroll
      for (int nt = 0; nt < 2; ++nt) {
        const unsigned short* base =
            &Bs[cur][(wn * 32 + nt * 16 + r) * LDS_STRIDE + ks * 32];
        b[nt].u[0] = *(const uint4*)(base + h * 8);
        b[nt].u[1] = *(const uint4*)(base + 16 + h * 8);
      }
      #pragma unroll
      for (int mt = 0; mt < 4; ++mt)
        #pragma unroll
        for (int nt = 0; nt < 2; ++nt)
          c[mt][nt] = __builtin_amdgcn_wmma_f32_16x16x32_bf16(
              false, a[mt].v, false, b[nt].v,
              (short)0, c[mt][nt], false, false);
    }

    // convert + store next chunk into the other buffer.
    // safe: last readers of buf (1-cur) finished at the previous barrier.
    if (kc + 1 < NCHUNK) stageChunk(1 - cur);

    __syncthreads();
  }

  // fold per-thread half-row norms into per-row totals (ds_add_f32)
  atomicAdd(&xsq_s[ldRow], accX);
  atomicAdd(&psq_s[ldRow], accP);
  __syncthreads();

  // epilogue: C layout = VGPR v <-> M = v + 8*(lane/16), N = lane%16
  // out = exp(-(xsq + psq - 2c)) = exp(2c - xsq - psq)   (GAMMA = 1)
  #pragma unroll
  for (int mt = 0; mt < 4; ++mt) {
    #pragma unroll
    for (int nt = 0; nt < 2; ++nt) {
      const int n  = wn * 32 + nt * 16 + r;
      const float pq = psq_s[n];
      float* outp = Out + (size_t)rowBase * Mcols + colBase + n;
      #pragma unroll
      for (int v = 0; v < 8; ++v) {
        const int m = wm * 64 + mt * 16 + 8 * h + v;
        outp[(size_t)m * Mcols] = __expf(2.0f * c[mt][nt][v] - xsq_s[m] - pq);
      }
    }
  }
}

extern "C" void kernel_launch(void* const* d_in, const int* in_sizes, int n_in,
                              void* d_out, int out_size, void* d_ws, size_t ws_size,
                              hipStream_t stream) {
  const float* x = (const float*)d_in[0];        // [N, 1024] f32
  const float* p = (const float*)d_in[1];        // [M, 1024] f32
  float* out = (float*)d_out;                    // [N, M] f32
  (void)d_ws; (void)ws_size; (void)n_in; (void)out_size;

  const int N = in_sizes[0] / K_DIM;             // 8192
  const int M = in_sizes[1] / K_DIM;             // 4096

  dim3 grid(M / TILE_N, N / TILE_M);             // (32, 64)
  rbf_gemm_wmma<<<grid, dim3(256), 0, stream>>>(x, p, out, M);
}
